// TransformerEncoder_22823456211230
// MI455X (gfx1250) — compile-verified, matrix operand order checked by reading, not run on hardware
//
#include <hip/hip_runtime.h>
#include <stdint.h>

#define DM 2048
#define NH 16
#define NG 4
#define HD 128
#define DF 8192
#define BB 2
#define TT 2048
#define NTOK (BB*TT)

typedef __bf16 v16bf __attribute__((ext_vector_type(16)));
typedef float  v8f   __attribute__((ext_vector_type(8)));

union FragBF { v16bf v; uint32_t u[8]; };

__device__ __forceinline__ uint16_t f2bf(float f) {
    uint32_t u = __float_as_uint(f);
    uint32_t r = (u + 0x7FFFu + ((u >> 16) & 1u)) >> 16;
    return (uint16_t)r;
}
__device__ __forceinline__ float bf2f(uint16_t h) {
    return __uint_as_float(((uint32_t)h) << 16);
}

// ---- CDNA5 async global->LDS copy (ASYNCcnt path), sync fallback ------------
#if __has_builtin(__builtin_amdgcn_global_load_async_to_lds_b128)
#define USE_ASYNC 1
#else
#define USE_ASYNC 0
#endif

typedef int v4i32 __attribute__((__vector_size__(4 * sizeof(int))));
typedef __attribute__((address_space(1))) v4i32 g_v4i32;
typedef __attribute__((address_space(3))) v4i32 l_v4i32;

__device__ __forceinline__ void cp16(uint16_t* l, const uint16_t* g) {
#if USE_ASYNC
    __builtin_amdgcn_global_load_async_to_lds_b128((g_v4i32*)g, (l_v4i32*)l, 0, 0);
#else
    *(uint4*)l = *(const uint4*)g;
#endif
}

template <int N>
__device__ __forceinline__ void wait_async() {
#if USE_ASYNC
#if __has_builtin(__builtin_amdgcn_s_wait_asynccnt)
    __builtin_amdgcn_s_wait_asynccnt(N);
#else
    asm volatile("s_wait_asynccnt %0" ::"i"(N));
#endif
#endif
}

// ---------------- weight convert + transpose: W[K][N] f32 -> Wt[N][K] bf16 ----
__global__ __launch_bounds__(256) void k_cvt_t(const float* __restrict__ W,
                                               uint16_t* __restrict__ Wt,
                                               int K, int N) {
    int idx = blockIdx.x * 256 + threadIdx.x;
    if (idx >= K * N) return;
    int k = idx / N, n = idx % N;
    Wt[(size_t)n * K + k] = f2bf(W[idx]);
}

// ---------------- RMSNorm -> bf16 --------------------------------------------
__global__ __launch_bounds__(256) void k_rmsnorm(const float* __restrict__ x,
                                                 const float* __restrict__ w,
                                                 uint16_t* __restrict__ out) {
    __shared__ float red[256];
    int row = blockIdx.x;
    const float* xr = x + (size_t)row * DM;
    float s = 0.f;
    for (int i = threadIdx.x; i < DM; i += 256) { float v = xr[i]; s += v * v; }
    red[threadIdx.x] = s; __syncthreads();
    for (int off = 128; off > 0; off >>= 1) {
        if (threadIdx.x < (unsigned)off) red[threadIdx.x] += red[threadIdx.x + off];
        __syncthreads();
    }
    float inv = rsqrtf(red[0] / (float)DM + 1e-6f);
    uint16_t* o = out + (size_t)row * DM;
    for (int i = threadIdx.x; i < DM; i += 256) o[i] = f2bf(xr[i] * inv * w[i]);
}

// ---------------- generic WMMA GEMM: C = [Res +] A(bf16)[M,K] @ Bt(bf16)[N,K]^T
// 128x128 block, 8 waves (2x4), 64x32 wave tile, double-buffered async LDS.
__global__ __launch_bounds__(256) void k_gemm(const uint16_t* __restrict__ A,
                                              const uint16_t* __restrict__ Bt,
                                              const float* __restrict__ Res,
                                              float* __restrict__ Cf,
                                              uint16_t* __restrict__ Cb,
                                              int M, int N, int K) {
    __shared__ __align__(16) uint16_t As[2][128][40];
    __shared__ __align__(16) uint16_t Bs[2][128][40];
    const int tid = threadIdx.x;
    const int mBase = blockIdx.y * 128, nBase = blockIdx.x * 128;
    const int wave = tid >> 5, lane = tid & 31, half = lane >> 4, l16 = lane & 15;
    const int wm = (wave & 1) * 64, wn = (wave >> 1) * 32;
    const int lr = tid >> 2, lc = (tid & 3) << 3;  // this thread's fixed LDS slot

    const uint16_t* gA0 = A  + (size_t)(mBase + lr) * K + lc;
    const uint16_t* gA1 = gA0 + (size_t)64 * K;
    const uint16_t* gB0 = Bt + (size_t)(nBase + lr) * K + lc;
    const uint16_t* gB1 = gB0 + (size_t)64 * K;

    const v8f vz = {0.f,0.f,0.f,0.f,0.f,0.f,0.f,0.f};
    v8f acc[4][2];
    for (int i = 0; i < 4; i++) for (int j = 0; j < 2; j++) acc[i][j] = vz;

    // prologue: stage tile 0 into buffer 0
    cp16(&As[0][lr][lc], gA0);  cp16(&As[0][lr + 64][lc], gA1);
    cp16(&Bs[0][lr][lc], gB0);  cp16(&Bs[0][lr + 64][lc], gB1);

    const int nk = K >> 5;
    for (int t = 0; t < nk; ++t) {
        const int cur = t & 1;
        if (t + 1 < nk) {
            const int kt = (t + 1) << 5;
            cp16(&As[cur ^ 1][lr][lc], gA0 + kt);  cp16(&As[cur ^ 1][lr + 64][lc], gA1 + kt);
            cp16(&Bs[cur ^ 1][lr][lc], gB0 + kt);  cp16(&Bs[cur ^ 1][lr + 64][lc], gB1 + kt);
            wait_async<4>();   // oldest 4 (tile t) have landed
        } else {
            wait_async<0>();
        }
        __syncthreads();

        FragBF a[4], b[2];
#pragma unroll
        for (int mi = 0; mi < 4; mi++) {
            int r = wm + mi * 16 + l16;
#pragma unroll
            for (int v = 0; v < 8; v++) {
                int k0 = ((v >> 2) << 4) + (half << 3) + ((v & 3) << 1);
                a[mi].u[v] = *(const uint32_t*)&As[cur][r][k0];
            }
        }
#pragma unroll
        for (int ni = 0; ni < 2; ni++) {
            int c = wn + ni * 16 + l16;
#pragma unroll
            for (int v = 0; v < 8; v++) {
                int k0 = ((v >> 2) << 4) + (half << 3) + ((v & 3) << 1);
                b[ni].u[v] = *(const uint32_t*)&Bs[cur][c][k0];
            }
        }
#pragma unroll
        for (int mi = 0; mi < 4; mi++)
#pragma unroll
            for (int ni = 0; ni < 2; ni++)
                acc[mi][ni] = __builtin_amdgcn_wmma_f32_16x16x32_bf16(
                    false, a[mi].v, false, b[ni].v, (short)0, acc[mi][ni], false, false);
        __syncthreads();
    }
#pragma unroll
    for (int mi = 0; mi < 4; mi++)
#pragma unroll
        for (int ni = 0; ni < 2; ni++) {
            int rb = mBase + wm + mi * 16 + (half << 3);
            int cb = nBase + wn + ni * 16 + l16;
#pragma unroll
            for (int v = 0; v < 8; v++) {
                size_t idx = (size_t)(rb + v) * N + cb;
                float val = acc[mi][ni][v];
                if (Res) val += Res[idx];
                if (Cf) Cf[idx] = val; else Cb[idx] = f2bf(val);
            }
        }
}

// ---------------- fused gate/up GEMM + SwiGLU -> bf16 ------------------------
// 128x64 block, 8 waves (4x2), 32x32 wave tile x 2 accumulator sets.
__global__ __launch_bounds__(256) void k_gemm_swiglu(const uint16_t* __restrict__ A,
                                                     const uint16_t* __restrict__ Bg,
                                                     const uint16_t* __restrict__ Bu,
                                                     uint16_t* __restrict__ Act,
                                                     int M, int N, int K) {
    __shared__ __align__(16) uint16_t As[2][128][40];
    __shared__ __align__(16) uint16_t Gs[2][64][40];
    __shared__ __align__(16) uint16_t Us[2][64][40];
    const int tid = threadIdx.x;
    const int mBase = blockIdx.y * 128, nBase = blockIdx.x * 64;
    const int wave = tid >> 5, lane = tid & 31, half = lane >> 4, l16 = lane & 15;
    const int wm = (wave & 3) * 32, wn = (wave >> 2) * 32;
    const int lr = tid >> 2, lc = (tid & 3) << 3;

    const uint16_t* gA0 = A  + (size_t)(mBase + lr) * K + lc;
    const uint16_t* gA1 = gA0 + (size_t)64 * K;
    const uint16_t* gG  = Bg + (size_t)(nBase + lr) * K + lc;
    const uint16_t* gU  = Bu + (size_t)(nBase + lr) * K + lc;

    const v8f vz = {0.f,0.f,0.f,0.f,0.f,0.f,0.f,0.f};
    v8f ag[2][2], au[2][2];
    for (int i = 0; i < 2; i++) for (int j = 0; j < 2; j++) { ag[i][j] = vz; au[i][j] = vz; }

    cp16(&As[0][lr][lc], gA0);  cp16(&As[0][lr + 64][lc], gA1);
    cp16(&Gs[0][lr][lc], gG);   cp16(&Us[0][lr][lc], gU);

    const int nk = K >> 5;
    for (int t = 0; t < nk; ++t) {
        const int cur = t & 1;
        if (t + 1 < nk) {
            const int kt = (t + 1) << 5;
            cp16(&As[cur ^ 1][lr][lc], gA0 + kt);  cp16(&As[cur ^ 1][lr + 64][lc], gA1 + kt);
            cp16(&Gs[cur ^ 1][lr][lc], gG + kt);   cp16(&Us[cur ^ 1][lr][lc], gU + kt);
            wait_async<4>();
        } else {
            wait_async<0>();
        }
        __syncthreads();

        FragBF a[2], bg[2], bu[2];
#pragma unroll
        for (int mi = 0; mi < 2; mi++) {
            int r = wm + mi * 16 + l16;
#pragma unroll
            for (int v = 0; v < 8; v++) {
                int k0 = ((v >> 2) << 4) + (half << 3) + ((v & 3) << 1);
                a[mi].u[v] = *(const uint32_t*)&As[cur][r][k0];
            }
        }
#pragma unroll
        for (int ni = 0; ni < 2; ni++) {
            int c = wn + ni * 16 + l16;
#pragma unroll
            for (int v = 0; v < 8; v++) {
                int k0 = ((v >> 2) << 4) + (half << 3) + ((v & 3) << 1);
                bg[ni].u[v] = *(const uint32_t*)&Gs[cur][c][k0];
                bu[ni].u[v] = *(const uint32_t*)&Us[cur][c][k0];
            }
        }
#pragma unroll
        for (int mi = 0; mi < 2; mi++)
#pragma unroll
            for (int ni = 0; ni < 2; ni++) {
                ag[mi][ni] = __builtin_amdgcn_wmma_f32_16x16x32_bf16(
                    false, a[mi].v, false, bg[ni].v, (short)0, ag[mi][ni], false, false);
                au[mi][ni] = __builtin_amdgcn_wmma_f32_16x16x32_bf16(
                    false, a[mi].v, false, bu[ni].v, (short)0, au[mi][ni], false, false);
            }
        __syncthreads();
    }
#pragma unroll
    for (int mi = 0; mi < 2; mi++)
#pragma unroll
        for (int ni = 0; ni < 2; ni++) {
            int rb = mBase + wm + mi * 16 + (half << 3);
            int cb = nBase + wn + ni * 16 + l16;
#pragma unroll
            for (int v = 0; v < 8; v++) {
                float g = ag[mi][ni][v], u = au[mi][ni][v];
                float s = g / (1.f + __expf(-g));
                Act[(size_t)(rb + v) * N + cb] = f2bf(s * u);
            }
        }
}

// ---------------- RoPE (bf16 in/out), layout [B*T][nh][128] ------------------
__global__ __launch_bounds__(256) void k_rope(const uint16_t* __restrict__ in,
                                              uint16_t* __restrict__ out, int nh) {
    int idx = blockIdx.x * 256 + threadIdx.x;
    int total = NTOK * nh * 64;
    if (idx >= total) return;
    int i = idx & 63;
    int h = (idx >> 6) % nh;
    int tok = idx / (64 * nh);
    int t = tok % TT;
    float freq = __powf(10000.0f, -(float)(2 * i) / 128.0f);
    float ang = (float)t * freq;
    float c = __cosf(ang), s = __sinf(ang);
    const uint16_t* p = in  + ((size_t)tok * nh + h) * HD;
    uint16_t*       o = out + ((size_t)tok * nh + h) * HD;
    float x1 = bf2f(p[i]), x2 = bf2f(p[i + 64]);
    o[i]      = f2bf(x1 * c - x2 * s);
    o[i + 64] = f2bf(x1 * s + x2 * c);
}

// ---------------- V transpose: [B][T][G][128] -> [B][G][128][T] --------------
__global__ __launch_bounds__(256) void k_vtrans(const uint16_t* __restrict__ v,
                                                uint16_t* __restrict__ vt) {
    int idx = blockIdx.x * 256 + threadIdx.x;
    if (idx >= NTOK * NG * HD) return;
    int d = idx & 127;
    int g = (idx >> 7) & 3;
    int tok = idx >> 9;
    int t = tok % TT, b = tok / TT;
    vt[(((size_t)b * NG + g) * HD + d) * TT + t] = v[idx];
}

// ---------------- sliding-window flash attention (1 wave / 16-query tile) ----
__global__ __launch_bounds__(32) void k_attn(const uint16_t* __restrict__ Q,
                                             const uint16_t* __restrict__ Kr,
                                             const uint16_t* __restrict__ Vt,
                                             uint16_t* __restrict__ Ctx,
                                             const int* __restrict__ pwl,
                                             const int* __restrict__ pwr) {
    const int qt = blockIdx.x, h = blockIdx.y, b = blockIdx.z;
    const int g = h / (NH / NG);
    const int lane = threadIdx.x, half = lane >> 4, l16 = lane & 15;
    const int wl = pwl[0], wr = pwr[0];
    const float scale = 0.08838834764831845f;  // 1/sqrt(128)
    __shared__ __align__(16) uint16_t Pl[16][40];

    FragBF qa[4];
    {
        const uint16_t* qp = Q + (((size_t)b * TT + qt * 16 + l16) * NH + h) * HD;
#pragma unroll
        for (int kc = 0; kc < 4; kc++)
#pragma unroll
            for (int v = 0; v < 8; v++) {
                int k0 = kc * 32 + ((v >> 2) << 4) + (half << 3) + ((v & 3) << 1);
                qa[kc].u[v] = *(const uint32_t*)&qp[k0];
            }
    }

    const v8f vz = {0.f,0.f,0.f,0.f,0.f,0.f,0.f,0.f};
    v8f acc[8];
#pragma unroll
    for (int nt = 0; nt < 8; nt++) acc[nt] = vz;
    float mrow[8], lrow[8];
#pragma unroll
    for (int v = 0; v < 8; v++) { mrow[v] = -1e30f; lrow[v] = 0.f; }

    int c0 = max(0, qt * 16 - wl) >> 5;
    int c1 = min((qt * 16 + 15 + wr) >> 5, (TT - 1) >> 5);

    for (int c = c0; c <= c1; ++c) {
        const int kb = c * 32;
        v8f s0 = vz, s1 = vz;
        {
            const uint16_t* kp0 = Kr + (((size_t)b * TT + kb + l16)      * NG + g) * HD;
            const uint16_t* kp1 = Kr + (((size_t)b * TT + kb + 16 + l16) * NG + g) * HD;
#pragma unroll
            for (int kc = 0; kc < 4; kc++) {
                FragBF b0, b1;
#pragma unroll
                for (int v = 0; v < 8; v++) {
                    int k0 = kc * 32 + ((v >> 2) << 4) + (half << 3) + ((v & 3) << 1);
                    b0.u[v] = *(const uint32_t*)&kp0[k0];
                    b1.u[v] = *(const uint32_t*)&kp1[k0];
                }
                s0 = __builtin_amdgcn_wmma_f32_16x16x32_bf16(false, qa[kc].v, false, b0.v, (short)0, s0, false, false);
                s1 = __builtin_amdgcn_wmma_f32_16x16x32_bf16(false, qa[kc].v, false, b1.v, (short)0, s1, false, false);
            }
        }
#pragma unroll
        for (int v = 0; v < 8; v++) {
            int m = v + (half << 3);
            int i = qt * 16 + m;
            int j0 = kb + l16, j1 = j0 + 16;
            bool a0 = (j0 >= i - wl) && (j0 <= i + wr);
            bool a1 = (j1 >= i - wl) && (j1 <= i + wr);
            float e0 = a0 ? s0[v] * scale : -1e30f;
            float e1 = a1 ? s1[v] * scale : -1e30f;
            float mx = fmaxf(e0, e1);
            for (int off = 1; off <= 8; off <<= 1) mx = fmaxf(mx, __shfl_xor(mx, off, 32));
            float nm = fmaxf(mrow[v], mx);
            float alpha = __expf(mrow[v] - nm);
            float p0 = a0 ? __expf(e0 - nm) : 0.f;
            float p1 = a1 ? __expf(e1 - nm) : 0.f;
            float sm = p0 + p1;
            for (int off = 1; off <= 8; off <<= 1) sm += __shfl_xor(sm, off, 32);
            lrow[v] = lrow[v] * alpha + sm;
            mrow[v] = nm;
#pragma unroll
            for (int nt = 0; nt < 8; nt++) acc[nt][v] *= alpha;
            Pl[m][l16]      = f2bf(p0);
            Pl[m][16 + l16] = f2bf(p1);
        }
        __syncthreads();
        FragBF pa;
#pragma unroll
        for (int v = 0; v < 8; v++) {
            int k0 = ((v >> 2) << 4) + (half << 3) + ((v & 3) << 1);
            pa.u[v] = *(const uint32_t*)&Pl[l16][k0];
        }
#pragma unroll
        for (int nt = 0; nt < 8; nt++) {
            FragBF vb;
            const uint16_t* vp = Vt + ((((size_t)b * NG + g) * HD) + nt * 16 + l16) * TT + kb;
#pragma unroll
            for (int v = 0; v < 8; v++) {
                int k0 = ((v >> 2) << 4) + (half << 3) + ((v & 3) << 1);
                vb.u[v] = *(const uint32_t*)&vp[k0];
            }
            acc[nt] = __builtin_amdgcn_wmma_f32_16x16x32_bf16(false, pa.v, false, vb.v, (short)0, acc[nt], false, false);
        }
        __syncthreads();
    }
#pragma unroll
    for (int v = 0; v < 8; v++) {
        int m = v + (half << 3);
        float inv = 1.f / lrow[v];
        size_t row = (size_t)(b * TT + qt * 16 + m) * DM + h * HD;
#pragma unroll
        for (int nt = 0; nt < 8; nt++)
            Ctx[row + nt * 16 + l16] = f2bf(acc[nt][v] * inv);
    }
}

// ---------------- host launcher ----------------------------------------------
extern "C" void kernel_launch(void* const* d_in, const int* in_sizes, int n_in,
                              void* d_out, int out_size, void* d_ws, size_t ws_size,
                              hipStream_t stream) {
    (void)in_sizes; (void)n_in; (void)out_size; (void)ws_size;
    const float* x   = (const float*)d_in[0];
    const float* wq  = (const float*)d_in[1];
    const float* wk  = (const float*)d_in[2];
    const float* wv  = (const float*)d_in[3];
    const float* wo  = (const float*)d_in[4];
    const float* anw = (const float*)d_in[5];
    const float* fnw = (const float*)d_in[6];
    const float* wg  = (const float*)d_in[7];
    const float* wu  = (const float*)d_in[8];
    const float* wd  = (const float*)d_in[9];
    const int* winl  = (const int*)d_in[10];
    const int* winr  = (const int*)d_in[11];
    float* out = (float*)d_out;

    char* ws = (char*)d_ws;
    size_t off = 0;
    auto take = [&](size_t bytes) -> char* {
        char* p = ws + off; off = (off + bytes + 255) & ~(size_t)255; return p;
    };
    uint16_t* wq_t = (uint16_t*)take((size_t)DM * DM * 2);
    uint16_t* wk_t = (uint16_t*)take((size_t)DM * NG * HD * 2);
    uint16_t* wv_t = (uint16_t*)take((size_t)DM * NG * HD * 2);
    uint16_t* wo_t = (uint16_t*)take((size_t)DM * DM * 2);
    uint16_t* wg_t = (uint16_t*)take((size_t)DM * DF * 2);
    uint16_t* wu_t = (uint16_t*)take((size_t)DM * DF * 2);
    uint16_t* wd_t = (uint16_t*)take((size_t)DF * DM * 2);
    float*    x2   = (float*)take((size_t)NTOK * DM * 4);
    size_t stage = off;
    uint16_t* h1  = (uint16_t*)take((size_t)NTOK * DM * 2);
    uint16_t* qbf = (uint16_t*)take((size_t)NTOK * NH * HD * 2);
    uint16_t* kbf = (uint16_t*)take((size_t)NTOK * NG * HD * 2);
    uint16_t* vbf = (uint16_t*)take((size_t)NTOK * NG * HD * 2);
    uint16_t* qr  = (uint16_t*)take((size_t)NTOK * NH * HD * 2);
    uint16_t* kr  = (uint16_t*)take((size_t)NTOK * NG * HD * 2);
    uint16_t* vt  = (uint16_t*)take((size_t)NTOK * NG * HD * 2);
    uint16_t* ctx = (uint16_t*)take((size_t)NTOK * DM * 2);
    off = stage;  // stage B overlays stage A (h1..ctx dead by then)
    uint16_t* h2  = (uint16_t*)take((size_t)NTOK * DM * 2);
    uint16_t* act = (uint16_t*)take((size_t)NTOK * DF * 2);

    k_cvt_t<<<(DM * DM + 255) / 256, 256, 0, stream>>>(wq, wq_t, DM, DM);
    k_cvt_t<<<(DM * NG * HD + 255) / 256, 256, 0, stream>>>(wk, wk_t, DM, NG * HD);
    k_cvt_t<<<(DM * NG * HD + 255) / 256, 256, 0, stream>>>(wv, wv_t, DM, NG * HD);
    k_cvt_t<<<(DM * DM + 255) / 256, 256, 0, stream>>>(wo, wo_t, DM, DM);
    k_cvt_t<<<(DM * DF + 255) / 256, 256, 0, stream>>>(wg, wg_t, DM, DF);
    k_cvt_t<<<(DM * DF + 255) / 256, 256, 0, stream>>>(wu, wu_t, DM, DF);
    k_cvt_t<<<(DF * DM + 255) / 256, 256, 0, stream>>>(wd, wd_t, DF, DM);
    k_rmsnorm<<<NTOK, 256, 0, stream>>>(x, anw, h1);
    k_gemm<<<dim3((NH * HD) / 128, NTOK / 128), 256, 0, stream>>>(h1, wq_t, nullptr, nullptr, qbf, NTOK, NH * HD, DM);
    k_gemm<<<dim3((NG * HD) / 128, NTOK / 128), 256, 0, stream>>>(h1, wk_t, nullptr, nullptr, kbf, NTOK, NG * HD, DM);
    k_gemm<<<dim3((NG * HD) / 128, NTOK / 128), 256, 0, stream>>>(h1, wv_t, nullptr, nullptr, vbf, NTOK, NG * HD, DM);
    k_rope<<<(NTOK * NH * 64 + 255) / 256, 256, 0, stream>>>(qbf, qr, NH);
    k_rope<<<(NTOK * NG * 64 + 255) / 256, 256, 0, stream>>>(kbf, kr, NG);
    k_vtrans<<<(NTOK * NG * HD + 255) / 256, 256, 0, stream>>>(vbf, vt);
    k_attn<<<dim3(TT / 16, NH, BB), 32, 0, stream>>>(qr, kr, vt, ctx, winl, winr);
    k_gemm<<<dim3(DM / 128, NTOK / 128), 256, 0, stream>>>(ctx, wo_t, x, x2, nullptr, NTOK, DM, DM);
    k_rmsnorm<<<NTOK, 256, 0, stream>>>(x2, fnw, h2);
    k_gemm_swiglu<<<dim3(DF / 64, NTOK / 128), 256, 0, stream>>>(h2, wg_t, wu_t, act, NTOK, DF, DM);
    k_gemm<<<dim3(DM / 128, NTOK / 128), 256, 0, stream>>>(act, wd_t, x2, out, nullptr, NTOK, DM, DF);
}
